// ReaRev_8263517078010
// MI455X (gfx1250) — matrix operand order repaired and checked by reading, not compiled, not run on hardware
//
#include <hip/hip_runtime.h>
#include <math.h>

// ---- problem constants (from reference) ----
#define B_   16
#define L_   32
#define QD_  300
#define N_   20000
#define E_   320000
#define ND_  100
#define RD_  200
#define H_   64
#define K_   2
#define NL_  2
#define NITER_ 2
#define KMAX 320   // max GEMM K (fuse: (2K+1)H = 320); all Kd, lda multiples of 4

typedef __attribute__((ext_vector_type(16))) _Float16 v16h;
typedef __attribute__((ext_vector_type(8)))  _Float16 v8h;
typedef __attribute__((ext_vector_type(8)))  float    v8f;

static __device__ __forceinline__ unsigned ordenc(float f) {
  unsigned u = __float_as_uint(f);
  return (u & 0x80000000u) ? ~u : (u | 0x80000000u);
}
static __device__ __forceinline__ float orddec(unsigned m) {
  return (m & 0x80000000u) ? __uint_as_float(m & 0x7fffffffu)
                           : __uint_as_float(~m);
}
static __device__ __forceinline__ int imin(int a, int b) { return a < b ? a : b; }

// ======================================================================
// Generic WMMA GEMM: Out[M,64] = act(A[M,Kd] @ W[Kd,64] + bias)
// block = 128 threads = 4 wave32; each wave computes a 16x64 tile.
// W staged fp32->fp16 in LDS in fragment-ready swizzled layout;
// A tiles staged per-wave with branch-free clamped float4 loads.
// Requires: Kd % 4 == 0, lda % 4 == 0 (true for all call sites).
// ======================================================================
__global__ __launch_bounds__(128)
void k_gemm64(const float* __restrict__ A, int M, int Kd, int lda,
              const float* __restrict__ W, const float* __restrict__ bias,
              float* __restrict__ Out, int relu_flag)
{
  // W fragments: [kchunk][ntile][lane][16 halves]  (10*4*32*16*2B = 40 KB)
  __shared__ _Float16 wlds[(KMAX / 32) * 4 * 32 * 16];
  __shared__ _Float16 alds[4][16 * 32];              // per-wave 16x32 tile, 4 KB
  const int tid  = threadIdx.x;
  const int wave = tid >> 5;
  const int lane = tid & 31;
  const int Kpad   = (Kd + 31) & ~31;
  const int nchunk = Kpad >> 5;

  // ---- stage W into fragment-ready layout (branch-free clamp+mask) ----
  // element (kcid, nt, ln, e)  <->  W[kcid*32 + e + (ln>>4)*16][nt*16 + (ln&15)]
  for (int idx = tid; idx < nchunk * 2048; idx += 128) {
    int e    = idx & 15;
    int ln   = (idx >> 4) & 31;
    int nt   = (idx >> 9) & 3;
    int kcid = idx >> 11;
    int kk   = kcid * 32 + e + ((ln >> 4) << 4);
    int col  = nt * 16 + (ln & 15);
    float v  = W[imin(kk, Kd - 1) * 64 + col];
    v = (kk < Kd) ? v : 0.0f;
    wlds[idx] = (_Float16)v;
  }
  __syncthreads();

  const int row0 = blockIdx.x * 64 + wave * 16;
  const int rsel = lane & 15;
  const int hi   = lane >> 4;

  // A staging geometry: lane stages row (lane&15), 16 cols at (lane>>4)*16
  const int arow_i = rsel;
  const int acol0  = hi * 16;
  const int grow_s = row0 + arow_i;
  const float* arow = A + (size_t)imin(grow_s, M - 1) * lda;
  const float rmask = (grow_s < M) ? 1.0f : 0.0f;

  v8f acc[4];
  #pragma unroll
  for (int nt = 0; nt < 4; ++nt) acc[nt] = (v8f){};

  for (int kcid = 0; kcid < nchunk; ++kcid) {
    const int kc = kcid << 5;
    // ---- stage A tile: 4x float4 loads, clamp+mask, pack f16, 2x b128 stores
    float vals[16];
    #pragma unroll
    for (int i = 0; i < 4; ++i) {
      int gk = kc + acol0 + i * 4;
      const float4 f = *(const float4*)(arow + imin(gk, Kd - 4));
      float m = (gk < Kd) ? rmask : 0.0f;
      vals[i * 4 + 0] = f.x * m;
      vals[i * 4 + 1] = f.y * m;
      vals[i * 4 + 2] = f.z * m;
      vals[i * 4 + 3] = f.w * m;
    }
    if (kcid + 1 < nchunk)
      __builtin_prefetch(arow + kc + 32 + acol0, 0, 0);   // global_prefetch_b8
    v8h h0, h1;
    #pragma unroll
    for (int j = 0; j < 8; ++j) {
      h0[j] = (_Float16)vals[j];
      h1[j] = (_Float16)vals[8 + j];
    }
    *(v8h*)&alds[wave][arow_i * 32 + acol0]     = h0;
    *(v8h*)&alds[wave][arow_i * 32 + acol0 + 8] = h1;
    // per-wave LDS RAW: hardware keeps DS ops of a wave in order.

    // ---- A fragment: two contiguous 16B runs in the row-major tile ----
    // a[0..7]  = tile[rsel][hi*16 .. hi*16+7]
    // a[8..15] = tile[rsel][hi*16 + 8 .. +7]
    v8h alo = *(const v8h*)&alds[wave][rsel * 32 + hi * 16];
    v8h ahi = *(const v8h*)&alds[wave][rsel * 32 + hi * 16 + 8];
    v16h a = __builtin_shufflevector(alo, ahi, 0, 1, 2, 3, 4, 5, 6, 7,
                                               8, 9, 10, 11, 12, 13, 14, 15);
    // ---- B fragments: contiguous 32B per lane in swizzled wlds ----
    #pragma unroll
    for (int nt = 0; nt < 4; ++nt) {
      const _Float16* wp = &wlds[((kcid * 4 + nt) * 32 + lane) * 16];
      v8h blo = *(const v8h*)wp;
      v8h bhi = *(const v8h*)(wp + 8);
      v16h b = __builtin_shufflevector(blo, bhi, 0, 1, 2, 3, 4, 5, 6, 7,
                                                 8, 9, 10, 11, 12, 13, 14, 15);
      acc[nt] = __builtin_amdgcn_wmma_f32_16x16x32_f16(
                    false, a, false, b, (short)0, acc[nt], false, false);
    }
  }

  // C/D layout: VGPR r -> row r (lanes 0-15) / row r+8 (lanes 16-31), col=lane&15
  const int rbase = hi * 8;
  #pragma unroll
  for (int r = 0; r < 8; ++r) {
    int grow = row0 + rbase + r;
    if (grow < M) {
      float v0 = acc[0][r] + bias[rsel];
      float v1 = acc[1][r] + bias[rsel + 16];
      float v2 = acc[2][r] + bias[rsel + 32];
      float v3 = acc[3][r] + bias[rsel + 48];
      if (relu_flag) {
        v0 = fmaxf(v0, 0.f); v1 = fmaxf(v1, 0.f);
        v2 = fmaxf(v2, 0.f); v3 = fmaxf(v3, 0.f);
      }
      float* orow = Out + (size_t)grow * 64;
      orow[rsel]      = v0; orow[rsel + 16] = v1;
      orow[rsel + 32] = v2; orow[rsel + 48] = v3;
    }
  }
}

// ======================================================================
// small utility kernels
// ======================================================================
__global__ void k_fillf(float* p, float v, int n) {
  int i = blockIdx.x * blockDim.x + threadIdx.x;
  if (i < n) p[i] = v;
}
__global__ void k_fillu(unsigned* p, unsigned v, int n) {
  int i = blockIdx.x * blockDim.x + threadIdx.x;
  if (i < n) p[i] = v;
}

// q_repr[b][h] = sum_l qt[b][l][h]*m[b][l] / max(sum_l m[b][l], 1e-9)
__global__ void k_qrepr(const float* __restrict__ qt, const float* __restrict__ qmask,
                        float* __restrict__ qrepr)
{
  int gid = blockIdx.x * blockDim.x + threadIdx.x;
  if (gid >= B_ * H_) return;
  int b = gid >> 6, h = gid & 63;
  float s = 0.f, ms = 0.f;
  for (int l = 0; l < L_; ++l) {
    float m = qmask[b * L_ + l];
    s += qt[(b * L_ + l) * H_ + h] * m;
    ms += m;
  }
  qrepr[gid] = s / fmaxf(ms, 1e-9f);
}

// Instruction generation: K attention steps over question tokens (1 block).
__global__ __launch_bounds__(256)
void k_instr(const float* __restrict__ qt, const float* __restrict__ qmask,
             const float* __restrict__ qrepr, const float* __restrict__ instr_init,
             const float* __restrict__ tokW, const float* __restrict__ tokb,
             const float* __restrict__ WkW, const float* __restrict__ Wkb,
             float* __restrict__ instr)
{
  __shared__ float prev[B_ * H_];
  __shared__ float qk[B_ * H_];
  __shared__ float sc[B_ * L_];
  int tid = threadIdx.x;
  for (int idx = tid; idx < B_ * H_; idx += 256)
    prev[idx] = instr_init[idx & 63];
  __syncthreads();

  for (int k = 0; k < K_; ++k) {
    // qk = ctx @ Wk[k] + bk ; ctx = [prev | q | q*prev | q-prev]  (B x 256)
    for (int idx = tid; idx < B_ * H_; idx += 256) {
      int b = idx >> 6, h = idx & 63;
      float acc = Wkb[k * H_ + h];
      for (int j = 0; j < 4 * H_; ++j) {
        int jj = j & 63;
        float pv = prev[b * H_ + jj];
        float qv = qrepr[b * H_ + jj];
        float c = (j < 64) ? pv : (j < 128) ? qv : (j < 192) ? qv * pv : qv - pv;
        acc += c * WkW[(size_t)k * 4 * H_ * H_ + j * H_ + h];
      }
      qk[idx] = acc;
    }
    __syncthreads();
    // scores[b][l]
    for (int idx = tid; idx < B_ * L_; idx += 256) {
      int b = idx >> 5, l = idx & 31;
      float acc = tokb[0];
      for (int h = 0; h < H_; ++h)
        acc += qk[b * H_ + h] * qt[(b * L_ + l) * H_ + h] * tokW[h];
      if (qmask[b * L_ + l] == 0.f) acc = -1e9f;
      sc[idx] = acc;
    }
    __syncthreads();
    // softmax over L per row
    if (tid < B_) {
      float mx = -3.4e38f;
      for (int l = 0; l < L_; ++l) mx = fmaxf(mx, sc[tid * L_ + l]);
      float s = 0.f;
      for (int l = 0; l < L_; ++l) {
        float e = expf(sc[tid * L_ + l] - mx);
        sc[tid * L_ + l] = e; s += e;
      }
      for (int l = 0; l < L_; ++l) sc[tid * L_ + l] /= s;
    }
    __syncthreads();
    // prev = w @ qt ; write instr[b][k][:]
    for (int idx = tid; idx < B_ * H_; idx += 256) {
      int b = idx >> 6, h = idx & 63;
      float a = 0.f;
      for (int l = 0; l < L_; ++l)
        a += sc[b * L_ + l] * qt[(b * L_ + l) * H_ + h];
      prev[idx] = a;
      instr[b * (K_ * H_) + k * H_ + h] = a;
    }
    __syncthreads();
  }
}

// p = where(seed,1,p); bsum[batch] += p   (LDS-binned)
__global__ void k_pseed(float* __restrict__ p, const unsigned char* __restrict__ seed,
                        const int* __restrict__ batch, float* __restrict__ bsum, int n)
{
  __shared__ float sb[B_];
  int tid = threadIdx.x;
  if (tid < B_) sb[tid] = 0.f;
  __syncthreads();
  int i = blockIdx.x * blockDim.x + tid;
  if (i < n) {
    float v = seed[i] ? 1.0f : p[i];
    p[i] = v;
    atomicAdd(&sb[batch[i]], v);
  }
  __syncthreads();
  if (tid < B_ && sb[tid] != 0.f) atomicAdd(&bsum[tid], sb[tid]);
}

__global__ void k_pdiv(float* __restrict__ p, const int* __restrict__ batch,
                       const float* __restrict__ bsum, float eps, int n)
{
  int i = blockIdx.x * blockDim.x + threadIdx.x;
  if (i < n) p[i] = p[i] / (bsum[batch[i]] + eps);
}

// concat[n][0:64]=xh[n], concat[n][64:320]=0
__global__ void k_concat_init(float* __restrict__ cc, const float* __restrict__ xh, int n)
{
  int gid = blockIdx.x * blockDim.x + threadIdx.x;
  if (gid >= n * 320) return;
  int row = gid / 320, c = gid % 320;
  cc[gid] = (c < 64) ? xh[row * 64 + c] : 0.0f;
}

// edge messages + scatter-add into concat aggr columns
__global__ void k_scatter(const float* __restrict__ ep, const float* __restrict__ instr,
                          const float* __restrict__ p, const int* __restrict__ src,
                          const int* __restrict__ dst, const int* __restrict__ etype,
                          const int* __restrict__ batch, float* __restrict__ cc)
{
  int gid = blockIdx.x * blockDim.x + threadIdx.x;
  if (gid >= E_ * H_) return;
  int e = gid >> 6, h = gid & 63;
  int d = dst[e];
  float pe = p[src[e]];
  float epv = ep[(size_t)e * H_ + h];
  int b = batch[d];
  int inv = (etype[e] == 1);
  int base = d * 320 + 64 + inv * (K_ * H_);
  #pragma unroll
  for (int k = 0; k < K_; ++k) {
    float m = instr[b * (K_ * H_) + k * H_ + h] * epv;
    m = fmaxf(m, 0.0f) * pe;
    atomicAdd(&cc[base + k * H_ + h], m);
  }
}

// logits[row] = dot(xh[row], probW)+probb; atomicMax bucket max (ordered uint)
__global__ __launch_bounds__(256)
void k_logits(const float* __restrict__ xh, const float* __restrict__ probW,
              const float* __restrict__ probb, const int* __restrict__ batch,
              float* __restrict__ logits, unsigned* __restrict__ bmax, int n)
{
  int row = blockIdx.x * 8 + (threadIdx.x >> 5);
  int lane = threadIdx.x & 31;
  if (row >= n) return;
  float s = xh[(size_t)row * 64 + lane]      * probW[lane]
          + xh[(size_t)row * 64 + lane + 32] * probW[lane + 32];
  for (int off = 16; off; off >>= 1) s += __shfl_xor(s, off, 32);
  if (lane == 0) {
    float lg = s + probb[0];
    logits[row] = lg;
    atomicMax(&bmax[batch[row]], ordenc(lg));
  }
}

// p[i] = exp(logits[i]-max); bsum[batch] += p[i]  (LDS-binned)
__global__ void k_expsum(const float* __restrict__ logits, const int* __restrict__ batch,
                         const unsigned* __restrict__ bmax, float* __restrict__ bsum,
                         float* __restrict__ p, int n)
{
  __shared__ float sb[B_];
  int tid = threadIdx.x;
  if (tid < B_) sb[tid] = 0.f;
  __syncthreads();
  int i = blockIdx.x * blockDim.x + tid;
  if (i < n) {
    float e = expf(logits[i] - orddec(bmax[batch[i]]));
    p[i] = e;
    atomicAdd(&sb[batch[i]], e);
  }
  __syncthreads();
  if (tid < B_ && sb[tid] != 0.f) atomicAdd(&bsum[tid], sb[tid]);
}

// brep[b][h] += xh[n][h]*p[n]   (LDS-binned 16x64)
__global__ void k_brep(const float* __restrict__ xh, const float* __restrict__ p,
                       const int* __restrict__ batch, float* __restrict__ brep, int n)
{
  __shared__ float sb[B_ * H_];
  int tid = threadIdx.x;
  for (int i = tid; i < B_ * H_; i += 256) sb[i] = 0.f;
  __syncthreads();
  int gid = blockIdx.x * 256 + tid;
  if (gid < n * H_) {
    int row = gid >> 6, h = gid & 63;
    atomicAdd(&sb[batch[row] * H_ + h], xh[gid] * p[row]);
  }
  __syncthreads();
  for (int i = tid; i < B_ * H_; i += 256)
    if (sb[i] != 0.f) atomicAdd(&brep[i], sb[i]);
}

// gated instruction update (1 block)
__global__ __launch_bounds__(256)
void k_gate(float* __restrict__ instr, const float* __restrict__ brep,
            const float* __restrict__ gateW, const float* __restrict__ resetW)
{
  __shared__ float newi[B_ * K_ * H_];
  int tid = threadIdx.x;
  for (int idx = tid; idx < B_ * K_ * H_; idx += 256) {
    int r = idx >> 6, h = idx & 63;
    int b = r >> 1, k = r & 1;
    float gacc = 0.f, racc = 0.f;
    for (int j = 0; j < 4 * H_; ++j) {
      int jj = j & 63;
      float iv = instr[b * (K_ * H_) + k * H_ + jj];
      float bv = brep[b * H_ + jj];
      // fusion = [instr, br, instr-br, instr*br]
      float f = (j < 64) ? iv : (j < 128) ? bv : (j < 192) ? iv - bv : iv * bv;
      gacc += f * gateW[j * H_ + h];
      racc += f * resetW[j * H_ + h];
    }
    float g = 1.f / (1.f + expf(-gacc));
    float self = instr[b * (K_ * H_) + k * H_ + h];
    newi[idx] = g * racc + (1.f - g) * self;
  }
  __syncthreads();
  for (int idx = tid; idx < B_ * K_ * H_; idx += 256) instr[idx] = newi[idx];
}

// ======================================================================
static inline int divup(int a, int b) { return (a + b - 1) / b; }

extern "C" void kernel_launch(void* const* d_in, const int* in_sizes, int n_in,
                              void* d_out, int out_size, void* d_ws, size_t ws_size,
                              hipStream_t stream)
{
  (void)in_sizes; (void)n_in; (void)out_size; (void)ws_size;
  const float* q_tokens  = (const float*)d_in[0];
  const float* q_mask    = (const float*)d_in[1];
  const float* x         = (const float*)d_in[2];
  const int*   eidx      = (const int*)d_in[3];
  const int*   etype     = (const int*)d_in[4];
  const float* edge_attr = (const float*)d_in[5];
  const unsigned char* seed = (const unsigned char*)d_in[6];
  const int*   batch     = (const int*)d_in[7];
  const float* node_in_W = (const float*)d_in[8];
  const float* node_in_b = (const float*)d_in[9];
  const float* q_W       = (const float*)d_in[10];
  const float* q_b       = (const float*)d_in[11];
  const float* instr_init= (const float*)d_in[12];
  const float* tok_W     = (const float*)d_in[13];
  const float* tok_b     = (const float*)d_in[14];
  const float* Wk_W      = (const float*)d_in[15];
  const float* Wk_b      = (const float*)d_in[16];
  const float* edge_W    = (const float*)d_in[17];
  const float* edge_b    = (const float*)d_in[18];
  const float* fuse_W    = (const float*)d_in[19];
  const float* fuse_b    = (const float*)d_in[20];
  const float* reset_W   = (const float*)d_in[21];
  const float* gate_W    = (const float*)d_in[22];
  const float* prob_W    = (const float*)d_in[23];
  const float* prob_b    = (const float*)d_in[24];

  const int* src = eidx;
  const int* dst = eidx + E_;
  float* p = (float*)d_out;   // output p lives directly in d_out (N floats)

  // --- workspace carve ---
  char* ws = (char*)d_ws;
  size_t off = 0;
  auto carve = [&](size_t bytes) -> char* {
    char* r = ws + off;
    off += (bytes + 255) & ~(size_t)255;
    return r;
  };
  float*    xh     = (float*)carve((size_t)N_ * H_ * 4);        // 5.1 MB
  float*    qt     = (float*)carve((size_t)B_ * L_ * H_ * 4);
  float*    qrepr  = (float*)carve((size_t)B_ * H_ * 4);
  float*    instr  = (float*)carve((size_t)B_ * K_ * H_ * 4);
  float*    ep     = (float*)carve((size_t)E_ * H_ * 4);        // 82 MB
  float*    cc     = (float*)carve((size_t)N_ * 320 * 4);       // 25.6 MB
  float*    logits = (float*)carve((size_t)N_ * 4);
  float*    bsum   = (float*)carve(B_ * 4);
  unsigned* bmax   = (unsigned*)carve(B_ * 4);
  float*    brep   = (float*)carve((size_t)B_ * H_ * 4);

  // 1. xh = relu(x @ node_in_W + b)       (20000x100 @ 100x64, WMMA)
  k_gemm64<<<divup(N_, 64), 128, 0, stream>>>(x, N_, ND_, ND_, node_in_W, node_in_b, xh, 1);
  // 2. qt = relu(q_tokens @ q_W + b)      (512x300 @ 300x64, WMMA)
  k_gemm64<<<divup(B_ * L_, 64), 128, 0, stream>>>(q_tokens, B_ * L_, QD_, QD_, q_W, q_b, qt, 1);
  // 3. masked mean of qt
  k_qrepr<<<divup(B_ * H_, 256), 256, 0, stream>>>(qt, q_mask, qrepr);
  // 4. instruction generation (K attention steps)
  k_instr<<<1, 256, 0, stream>>>(qt, q_mask, qrepr, instr_init, tok_W, tok_b, Wk_W, Wk_b, instr);
  // 5. p = 0
  k_fillf<<<divup(N_, 256), 256, 0, stream>>>(p, 0.0f, N_);

  for (int t = 0; t < NITER_; ++t) {
    // p = where(seed,1,p); p /= segsum(p)+1e-9
    k_fillf<<<1, 32, 0, stream>>>(bsum, 0.0f, B_);
    k_pseed<<<divup(N_, 256), 256, 0, stream>>>(p, seed, batch, bsum, N_);
    k_pdiv<<<divup(N_, 256), 256, 0, stream>>>(p, batch, bsum, 1e-9f, N_);

    for (int l = 0; l < NL_; ++l) {
      // ep = edge_attr @ edge_W[l] + edge_b[l]   (320000x200 @ 200x64, WMMA)
      k_gemm64<<<divup(E_, 64), 128, 0, stream>>>(
          edge_attr, E_, RD_, RD_,
          edge_W + (size_t)l * RD_ * H_, edge_b + l * H_, ep, 0);
      // concat = [xh | 0 | 0]
      k_concat_init<<<divup(N_ * 320, 256), 256, 0, stream>>>(cc, xh, N_);
      // messages + scatter-add into aggr columns
      k_scatter<<<divup(E_ * H_, 256), 256, 0, stream>>>(
          ep, instr, p, src, dst, etype, batch, cc);
      // xh = relu(concat @ fuse_W[l] + fuse_b[l])  (20000x320 @ 320x64, WMMA)
      k_gemm64<<<divup(N_, 64), 128, 0, stream>>>(
          cc, N_, 320, 320,
          fuse_W + (size_t)l * 320 * H_, fuse_b + l * H_, xh, 1);
      // per-graph softmax over logits
      k_fillu<<<1, 32, 0, stream>>>(bmax, 0u, B_);
      k_fillf<<<1, 32, 0, stream>>>(bsum, 0.0f, B_);
      k_logits<<<divup(N_, 8), 256, 0, stream>>>(xh, prob_W, prob_b, batch, logits, bmax, N_);
      k_expsum<<<divup(N_, 256), 256, 0, stream>>>(logits, batch, bmax, bsum, p, N_);
      k_pdiv<<<divup(N_, 256), 256, 0, stream>>>(p, batch, bsum, 1e-16f, N_);
    }

    if (t < NITER_ - 1) {
      k_fillf<<<divup(B_ * H_, 256), 256, 0, stream>>>(brep, 0.0f, B_ * H_);
      k_brep<<<divup(N_ * H_, 256), 256, 0, stream>>>(xh, p, batch, brep, N_);
      k_gate<<<1, 256, 0, stream>>>(instr, brep, gate_W, reset_W);
    }
  }
}
